// GLTCH_9655086481997
// MI455X (gfx1250) — matrix operand order repaired and checked by reading, not compiled, not run on hardware
//
#include <hip/hip_runtime.h>
#include <cstdint>
#include <cstddef>

typedef __attribute__((ext_vector_type(16))) __bf16 v16bf;
typedef __attribute__((ext_vector_type(8)))  __bf16 v8bf;
typedef __attribute__((ext_vector_type(8)))  float  v8f;

constexpr int B_  = 2;
constexpr int T_  = 1024;
constexpr int D_  = 1024;
constexpr int H_  = 16;
constexpr int HD_ = 64;
constexpr int L_  = 8;
constexpr int FF_ = 4096;
constexpr int V_  = 32000;
constexpr int M_  = B_ * T_;   // 2048 token rows

enum {
  FLAG_RES  = 1,
  FLAG_GELU = 2,
  FLAG_OUTF = 4,
  FLAG_OUTB = 8,
  FLAG_OUTT = 16,
  FLAG_BIAS = 32
};

// ---------- helpers ----------
__device__ __forceinline__ __bf16 f2bf(float f) {
  uint32_t u = __builtin_bit_cast(uint32_t, f);
  u += 0x7FFFu + ((u >> 16) & 1u);               // round to nearest even
  uint16_t h = (uint16_t)(u >> 16);
  return __builtin_bit_cast(__bf16, h);
}

// Async 16B global->LDS copy (GLOBAL_LOAD_ASYNC_TO_LDS_B128, ASYNCcnt-tracked).
// LDS destination: low 32 bits of the flat pointer == LDS offset.
__device__ __forceinline__ void async_copy16(const void* gsrc, void* lds_dst) {
  uint32_t l = (uint32_t)(uintptr_t)lds_dst;
  asm volatile("global_load_async_to_lds_b128 %0, %1, off"
               :: "v"(l), "v"(gsrc) : "memory");
}
__device__ __forceinline__ void async_wait0() {
  asm volatile("s_wait_asynccnt 0x0" ::: "memory");
}

#define CAT16(lo, hi) __builtin_shufflevector((lo), (hi), 0, 1, 2, 3, 4, 5, 6, 7, \
                                              8, 9, 10, 11, 12, 13, 14, 15)

// A fragment (16x32 bf16, ISA 7.12.2): lane row = lane&15; per half-wave the 16
// K values are two contiguous runs of 8: [khalf*8 .. +8) and [16+khalf*8 .. +8).
__device__ __forceinline__ v16bf frag_a(const __bf16* rowp, int khalf) {
  v8bf lo = *reinterpret_cast<const v8bf*>(rowp + (khalf << 3));
  v8bf hi = *reinterpret_cast<const v8bf*>(rowp + 16 + (khalf << 3));
  return CAT16(lo, hi);
}
// B fragment (32x16 bf16): lane col = lane&15; K values are two contiguous runs
// of 8: [khalf*8 .. +8) and [16+khalf*8 .. +8) when LDS is stored [col][k].
__device__ __forceinline__ v16bf frag_b(const __bf16* colp, int khalf) {
  v8bf lo = *reinterpret_cast<const v8bf*>(colp + (khalf << 3));
  v8bf hi = *reinterpret_cast<const v8bf*>(colp + 16 + (khalf << 3));
  return CAT16(lo, hi);
}

// ---------- embedding: x = tok_emb[idx] + pos_emb ----------
__global__ __launch_bounds__(256) void k_embed(const int* __restrict__ idx,
                                               const float* __restrict__ tok,
                                               const float* __restrict__ pos,
                                               float* __restrict__ x) {
  int row = blockIdx.x;                 // 0..M_-1  (= b*T + t)
  int t   = row % T_;
  int v   = idx[row];
  const float* te = tok + (size_t)v * D_;
  const float* pe = pos + (size_t)t * D_;
  float* xo = x + (size_t)row * D_;
  for (int d = threadIdx.x; d < D_; d += 256) xo[d] = te[d] + pe[d];
}

// ---------- weight convert + transpose: W f32[K,N] -> Wt bf16[N,K] ----------
__global__ __launch_bounds__(256) void k_cvtT(const float* __restrict__ W,
                                              uint16_t* __restrict__ Wt_,
                                              int K, int N) {
  __shared__ float tile[32][33];
  __bf16* Wt = reinterpret_cast<__bf16*>(Wt_);
  int bk = blockIdx.y * 32, bn = blockIdx.x * 32;
  int tx = threadIdx.x & 31, ty = threadIdx.x >> 5;
#pragma unroll
  for (int i = 0; i < 4; ++i)
    tile[ty + i * 8][tx] = W[(size_t)(bk + ty + i * 8) * N + bn + tx];
  __syncthreads();
#pragma unroll
  for (int i = 0; i < 4; ++i)
    Wt[(size_t)(bn + ty + i * 8) * K + bk + tx] = f2bf(tile[tx][ty + i * 8]);
}

// ---------- LayerNorm -> bf16 activations ----------
__global__ __launch_bounds__(256) void k_ln(const float* __restrict__ x,
                                            const float* __restrict__ sc,
                                            const float* __restrict__ sh,
                                            uint16_t* __restrict__ out_) {
  __bf16* out = reinterpret_cast<__bf16*>(out_);
  int row = blockIdx.x;
  const float* xr = x + (size_t)row * D_;
  float vals[4];
  float lsum = 0.f, lsq = 0.f;
#pragma unroll
  for (int i = 0; i < 4; ++i) {
    float t = xr[threadIdx.x + i * 256];
    vals[i] = t; lsum += t; lsq += t * t;
  }
#pragma unroll
  for (int d = 1; d < 32; d <<= 1) {
    lsum += __shfl_xor(lsum, d, 32);
    lsq  += __shfl_xor(lsq,  d, 32);
  }
  __shared__ float ss[8], sq[8];
  int wave = threadIdx.x >> 5, lane = threadIdx.x & 31;
  if (lane == 0) { ss[wave] = lsum; sq[wave] = lsq; }
  __syncthreads();
  if (wave == 0) {
    float a = (lane < 8) ? ss[lane] : 0.f;
    float b = (lane < 8) ? sq[lane] : 0.f;
#pragma unroll
    for (int d = 1; d < 8; d <<= 1) {
      a += __shfl_xor(a, d, 32);
      b += __shfl_xor(b, d, 32);
    }
    if (lane == 0) { ss[0] = a; sq[0] = b; }
  }
  __syncthreads();
  float mean = ss[0] * (1.f / D_);
  float var  = sq[0] * (1.f / D_) - mean * mean;
  float rstd = rsqrtf(var + 1e-5f);
#pragma unroll
  for (int i = 0; i < 4; ++i) {
    int d = threadIdx.x + i * 256;
    out[(size_t)row * D_ + d] = f2bf((vals[i] - mean) * rstd * sc[d] + sh[d]);
  }
}

// ---------- WMMA GEMM: C[M,N] = A_bf16[M,K] @ Wt_bf16[N,K]^T (+bias/...) ----
// 64x128 block tile, 8 waves each owning a 32x32 patch (2x2 WMMA tiles).
// Double-buffered async global->LDS staging overlapped with WMMA.
template <int FLAGS>
__global__ __launch_bounds__(256) void k_gemm(const uint16_t* __restrict__ Abf,
                                              const uint16_t* __restrict__ Wtb,
                                              const float* __restrict__ bias,
                                              const float* __restrict__ res,
                                              float* __restrict__ outf,
                                              uint16_t* __restrict__ outb,
                                              int M, int N, int K) {
  __shared__ alignas(16) __bf16 sA[2][64 * 32];    // [row][k]
  __shared__ alignas(16) __bf16 sBt[2][128 * 32];  // [col][k]  (K-contiguous)
  const __bf16* A  = reinterpret_cast<const __bf16*>(Abf);
  const __bf16* Bt = reinterpret_cast<const __bf16*>(Wtb);
  int tid = threadIdx.x, lane = tid & 31, wave = tid >> 5;
  int l15 = lane & 15, khalf = lane >> 4;
  int wm = wave & 1, wn = wave >> 1;               // 2 x 4 wave grid
  int bm = blockIdx.y * 64, bn = blockIdx.x * 128;

  v8f acc[2][2] = {};

  // staging thread mappings (16B chunks)
  int ar = tid >> 2, ac = (tid & 3) * 8;           // A: 1 chunk / thread
  int bc = tid & 127, bk = (tid >> 7) * 16;        // B: 2 chunks / thread
  const __bf16* ag = A  + (size_t)(bm + ar) * K + ac;
  const __bf16* bg = Bt + (size_t)(bn + bc) * K + bk;

  auto stage = [&](int buf, int k0) {
    async_copy16(ag + k0,     &sA[buf][ar * 32 + ac]);
    async_copy16(bg + k0,     &sBt[buf][bc * 32 + bk]);
    async_copy16(bg + k0 + 8, &sBt[buf][bc * 32 + bk + 8]);
  };

  int nk = K >> 5;
  stage(0, 0);
  for (int i = 0; i < nk; ++i) {
    int cur = i & 1;
    async_wait0();        // current buffer's copies complete (ours)
    __syncthreads();      // ... and everyone else's; prior reads also done
    if (i + 1 < nk) stage(cur ^ 1, (i + 1) << 5);

    v16bf af[2], bf[2];
#pragma unroll
    for (int mt = 0; mt < 2; ++mt)
      af[mt] = frag_a(&sA[cur][((wm * 2 + mt) * 16 + l15) * 32], khalf);
#pragma unroll
    for (int nt = 0; nt < 2; ++nt)
      bf[nt] = frag_b(&sBt[cur][((wn * 2 + nt) * 16 + l15) * 32], khalf);
#pragma unroll
    for (int mt = 0; mt < 2; ++mt)
#pragma unroll
      for (int nt = 0; nt < 2; ++nt)
        acc[mt][nt] = __builtin_amdgcn_wmma_f32_16x16x32_bf16(
            false, af[mt], false, bf[nt], (short)0, acc[mt][nt], false, false);
  }

  // epilogue: C layout row = j + 8*khalf, col = lane&15
#pragma unroll
  for (int mt = 0; mt < 2; ++mt) {
#pragma unroll
    for (int nt = 0; nt < 2; ++nt) {
#pragma unroll
      for (int j = 0; j < 8; ++j) {
        int r = bm + (wm * 2 + mt) * 16 + j + (khalf << 3);
        int c = bn + (wn * 2 + nt) * 16 + l15;
        float v = acc[mt][nt][j];
        if (FLAGS & FLAG_BIAS) v += bias[c];
        size_t o = (size_t)r * N + c;
        if (FLAGS & FLAG_RES)  v += res[o];
        if (FLAGS & FLAG_GELU) v = 0.5f * v * (1.0f + erff(v * 0.7071067811865475f));
        if (FLAGS & FLAG_OUTF) outf[o] = v;
        if (FLAGS & FLAG_OUTB) reinterpret_cast<__bf16*>(outb)[o] = f2bf(v);
        if (FLAGS & FLAG_OUTT) {   // per-head transposed bf16: [B*H][HD][T]
          int t = r % T_, bb = r / T_;
          int hh = c >> 6, d = c & 63;
          reinterpret_cast<__bf16*>(outb)[((size_t)(bb * H_ + hh) * HD_ + d) * T_ + t]
              = f2bf(v);
        }
      }
    }
  }
}

// ---------- flash attention (causal), WMMA for QK^T and PV ----------
// q,k: bf16 [M][D];  vT: bf16 [B*H][HD][T]; double-buffered async staging.
// Each wave owns 16 q-rows -> causal skip/edge handling is wave-uniform.
__global__ __launch_bounds__(256) void k_attn(const uint16_t* __restrict__ q_,
                                              const uint16_t* __restrict__ k_,
                                              const uint16_t* __restrict__ vT_,
                                              uint16_t* __restrict__ o_) {
  __shared__ alignas(16) __bf16 sQ[128 * 64];       // [row][dim]
  __shared__ alignas(16) __bf16 sK[2][32 * 64];     // [key][dim]
  __shared__ alignas(16) __bf16 sVt[2][64 * 32];    // [dim][key]
  __shared__ alignas(16) __bf16 sP[8][16 * 32];     // per-wave P relayout
  const __bf16* q  = reinterpret_cast<const __bf16*>(q_);
  const __bf16* kk = reinterpret_cast<const __bf16*>(k_);
  const __bf16* vT = reinterpret_cast<const __bf16*>(vT_);
  __bf16* o = reinterpret_cast<__bf16*>(o_);

  int bh = blockIdx.y, b = bh / H_, h = bh % H_;
  int qbase = blockIdx.x * 128;
  int tid = threadIdx.x, lane = tid & 31, wave = tid >> 5;
  int l15 = lane & 15, khalf = lane >> 4;

  int kr = tid >> 3, kd = (tid & 7) * 8;   // K tile chunk
  int vd = tid >> 2, vc = (tid & 3) * 8;   // V tile chunk (already transposed)
  const __bf16* kg = kk + ((size_t)(b * T_ + kr)) * D_ + h * HD_ + kd;
  const __bf16* vg = vT + ((size_t)bh * HD_ + vd) * T_ + vc;

  auto stageKV = [&](int buf, int kb) {
    async_copy16(kg + (size_t)kb * D_, &sK[buf][kr * 64 + kd]);
    async_copy16(vg + kb,              &sVt[buf][vd * 32 + vc]);
  };

  // stage Q: 4 async 16B chunks per thread (128 rows x 64 dims)
#pragma unroll
  for (int g = 0; g < 4; ++g) {
    int i = tid + g * 256;
    int r = i >> 3, d8 = (i & 7) * 8;
    async_copy16(q + ((size_t)(b * T_ + qbase + r)) * D_ + h * HD_ + d8,
                 &sQ[r * 64 + d8]);
  }
  stageKV(0, 0);

  float rm[8], rl[8];
  v8f zero = {};
  v8f oacc[4];
#pragma unroll
  for (int j = 0; j < 8; ++j) { rm[j] = -1e30f; rl[j] = 0.f; }
#pragma unroll
  for (int n = 0; n < 4; ++n) oacc[n] = zero;

  const float scale = 0.125f;  // 1/sqrt(64)
  int nkb = (qbase + 128) >> 5;   // causal: key blocks 0 .. qbase+127
  int rowlo = qbase + wave * 16;  // this wave's q-row range (uniform)
  int rowhi = rowlo + 15;

  for (int i = 0; i < nkb; ++i) {
    int cur = i & 1, kb = i << 5;
    async_wait0();
    __syncthreads();
    if (i + 1 < nkb) stageKV(cur ^ 1, (i + 1) << 5);

    if (kb > rowhi) continue;          // fully masked for this wave: skip
    bool edge = (kb + 31 > rowlo);     // diagonal block: needs element mask

    // S = Q(16x64) @ K^T(64x32)
    v8f s0 = zero, s1 = zero;
#pragma unroll
    for (int ks = 0; ks < 2; ++ks) {
      v16bf aq = frag_a(&sQ[(wave * 16 + l15) * 64 + ks * 32], khalf);
      v16bf b0 = frag_b(&sK[cur][l15 * 64 + ks * 32], khalf);
      v16bf b1 = frag_b(&sK[cur][(16 + l15) * 64 + ks * 32], khalf);
      s0 = __builtin_amdgcn_wmma_f32_16x16x32_bf16(false, aq, false, b0,
                                                   (short)0, s0, false, false);
      s1 = __builtin_amdgcn_wmma_f32_16x16x32_bf16(false, aq, false, b1,
                                                   (short)0, s1, false, false);
    }

    // online softmax, wave-local per row (16-lane half reductions)
#pragma unroll
    for (int j = 0; j < 8; ++j) {
      float v0 = s0[j] * scale;
      float v1 = s1[j] * scale;
      if (edge) {                      // wave-uniform branch
        int rowg = rowlo + j + (khalf << 3);
        if (kb + l15      > rowg) v0 = -1e30f;
        if (kb + 16 + l15 > rowg) v1 = -1e30f;
      }
      float mx = fmaxf(v0, v1);
#pragma unroll
      for (int d = 1; d < 16; d <<= 1) mx = fmaxf(mx, __shfl_xor(mx, d, 32));
      float mnew = fmaxf(rm[j], mx);
      float corr = __expf(rm[j] - mnew);
      float p0 = __expf(v0 - mnew);
      float p1 = __expf(v1 - mnew);
      float rs = p0 + p1;
#pragma unroll
      for (int d = 1; d < 16; d <<= 1) rs += __shfl_xor(rs, d, 32);
      rl[j] = rl[j] * corr + rs;
      rm[j] = mnew;
#pragma unroll
      for (int n = 0; n < 4; ++n) oacc[n][j] *= corr;
      sP[wave][(j + (khalf << 3)) * 32 + l15]      = f2bf(p0);
      sP[wave][(j + (khalf << 3)) * 32 + 16 + l15] = f2bf(p1);
    }

    // O += P(16x32) @ V(32x64)   (sP traffic is wave-local, in-order)
    v16bf ap = frag_a(&sP[wave][l15 * 32], khalf);
#pragma unroll
    for (int n = 0; n < 4; ++n) {
      v16bf bv = frag_b(&sVt[cur][(n * 16 + l15) * 32], khalf);
      oacc[n] = __builtin_amdgcn_wmma_f32_16x16x32_bf16(false, ap, false, bv,
                                                        (short)0, oacc[n],
                                                        false, false);
    }
  }

#pragma unroll
  for (int j = 0; j < 8; ++j) {
    float inv = 1.f / rl[j];
    int r = qbase + wave * 16 + j + (khalf << 3);
#pragma unroll
    for (int n = 0; n < 4; ++n)
      o[((size_t)(b * T_ + r)) * D_ + h * HD_ + n * 16 + l15] =
          f2bf(oacc[n][j] * inv);
  }
}

// ---------- driver ----------
extern "C" void kernel_launch(void* const* d_in, const int* in_sizes, int n_in,
                              void* d_out, int out_size, void* d_ws, size_t ws_size,
                              hipStream_t stream) {
  (void)in_sizes; (void)n_in; (void)out_size; (void)ws_size;
  const int*   idx  = (const int*)  d_in[0];
  const float* tok  = (const float*)d_in[1];
  const float* pos  = (const float*)d_in[2];
  const float* Wq   = (const float*)d_in[3];
  const float* Wk   = (const float*)d_in[4];
  const float* Wv   = (const float*)d_in[5];
  const float* Wp   = (const float*)d_in[6];
  const float* bp   = (const float*)d_in[7];
  const float* ln1s = (const float*)d_in[8];
  const float* ln1b = (const float*)d_in[9];
  const float* W1   = (const float*)d_in[10];
  const float* b1   = (const float*)d_in[11];
  const float* W2   = (const float*)d_in[12];
  const float* b2   = (const float*)d_in[13];
  const float* ln2s = (const float*)d_in[14];
  const float* ln2b = (const float*)d_in[15];
  const float* lnfs = (const float*)d_in[16];
  const float* lnfb = (const float*)d_in[17];
  const float* Wh   = (const float*)d_in[18];
  const float* bh   = (const float*)d_in[19];
  float* out = (float*)d_out;

  char* ws = (char*)d_ws;
  size_t off = 0;
  auto carve = [&](size_t bytes) -> void* {
    void* p = ws + off;
    off += (bytes + 255) & ~(size_t)255;
    return p;
  };
  float*    x   = (float*)   carve((size_t)M_ * D_  * 4);
  uint16_t* qbf = (uint16_t*)carve((size_t)M_ * D_  * 2);
  uint16_t* kbf = (uint16_t*)carve((size_t)M_ * D_  * 2);
  uint16_t* vtb = (uint16_t*)carve((size_t)M_ * D_  * 2);  // [B*H][HD][T]
  uint16_t* hbf = (uint16_t*)carve((size_t)M_ * D_  * 2);
  uint16_t* obf = (uint16_t*)carve((size_t)M_ * D_  * 2);
  uint16_t* ubf = (uint16_t*)carve((size_t)M_ * FF_ * 2);
  uint16_t* wtb = (uint16_t*)carve((size_t)D_ * V_  * 2);  // bf16 [N][K] slab

  dim3 blk(256);
  dim3 gD(D_ / 128, M_ / 64);    // N=1024 GEMMs
  auto cvt = [&](const float* W, int K, int N) {
    k_cvtT<<<dim3(N / 32, K / 32), blk, 0, stream>>>(W, wtb, K, N);
  };

  k_embed<<<dim3(M_), blk, 0, stream>>>(idx, tok, pos, x);

  for (int l = 0; l < L_; ++l) {
    size_t wo = (size_t)l * D_ * D_;
    k_ln<<<dim3(M_), blk, 0, stream>>>(x, ln1s + (size_t)l * D_, ln1b + (size_t)l * D_, hbf);
    cvt(Wq + wo, D_, D_);
    k_gemm<FLAG_OUTB><<<gD, blk, 0, stream>>>(hbf, wtb, nullptr, nullptr, nullptr, qbf,
                                              M_, D_, D_);
    cvt(Wk + wo, D_, D_);
    k_gemm<FLAG_OUTB><<<gD, blk, 0, stream>>>(hbf, wtb, nullptr, nullptr, nullptr, kbf,
                                              M_, D_, D_);
    cvt(Wv + wo, D_, D_);
    k_gemm<FLAG_OUTT><<<gD, blk, 0, stream>>>(hbf, wtb, nullptr, nullptr, nullptr, vtb,
                                              M_, D_, D_);
    k_attn<<<dim3(T_ / 128, B_ * H_), blk, 0, stream>>>(qbf, kbf, vtb, obf);
    cvt(Wp + wo, D_, D_);
    k_gemm<FLAG_BIAS | FLAG_RES | FLAG_OUTF><<<gD, blk, 0, stream>>>(
        obf, wtb, bp + (size_t)l * D_, x, x, nullptr, M_, D_, D_);
    k_ln<<<dim3(M_), blk, 0, stream>>>(x, ln2s + (size_t)l * D_, ln2b + (size_t)l * D_, hbf);
    cvt(W1 + (size_t)l * D_ * FF_, D_, FF_);
    k_gemm<FLAG_BIAS | FLAG_GELU | FLAG_OUTB><<<dim3(FF_ / 128, M_ / 64), blk, 0, stream>>>(
        hbf, wtb, b1 + (size_t)l * FF_, nullptr, nullptr, ubf, M_, FF_, D_);
    cvt(W2 + (size_t)l * FF_ * D_, FF_, D_);
    k_gemm<FLAG_BIAS | FLAG_RES | FLAG_OUTF><<<gD, blk, 0, stream>>>(
        ubf, wtb, b2 + (size_t)l * D_, x, x, nullptr, M_, D_, FF_);
  }

  k_ln<<<dim3(M_), blk, 0, stream>>>(x, lnfs, lnfb, hbf);
  cvt(Wh, D_, V_);
  k_gemm<FLAG_BIAS | FLAG_OUTF><<<dim3(V_ / 128, M_ / 64), blk, 0, stream>>>(
      hbf, wtb, bh, nullptr, out, nullptr, M_, V_, D_);
}
